// FlagOS_ChunkGatedDelta_6803228197380
// MI455X (gfx1250) — matrix operand order repaired and checked by reading, not compiled, not run on hardware
//
#include <hip/hip_runtime.h>

// Chunked gated-delta-rule forward for MI455X (gfx1250), wave32 + WMMA bf16.
// Shapes: B=4, H=8, S=4096, D=128, BT=64. Grid = (B*H)*4 e-slices, block = 128 (4 waves).

#define BT   64
#define DDIM 128
#define NCH  64
#define ESL  4
#define EW   32      // e-slice width
#define EPSF 1e-6f

// padded LDS strides (bf16 shorts / f32 words); chosen so vectorized stores are
// 8B-aligned and fragment loads hit distinct banks.
#define QS   132     // 128 + 4 (264B rows: 8B aligned, bank = 2m + d/2)
#define KTS  68      // 64 + 4  (136B rows: 8B aligned, bank = 34d + i/2)
#define VTS  66      // 64 + 2
#define AS   66
#define STS  132     // f32 state stride: 128 + 4
#define SBS  130     // bf16 state stride: 128 + 2

typedef __bf16 bf16;
typedef __attribute__((ext_vector_type(16))) __bf16 bf16x16;
typedef __attribute__((ext_vector_type(8)))  float  f32x8;

union ABFrag {
    bf16x16      bf;
    unsigned int u32[8];
};

union Pack4 {
    bf16  h[4];
    uint2 u2;
};

// A-fragment (16x32 bf16, M x K): lane m = lane&15, h = lane>>4.
// reg r holds K = 2*(r&3) + 8*h + 16*(r>>2) (+1). Source row-major [row][k].
__device__ __forceinline__ void load_a_frag(ABFrag& f, const bf16* base,
                                            int row0, int k0, int stride, int n, int h) {
#pragma unroll
    for (int r = 0; r < 8; ++r) {
        int kl = ((r & 3) * 2) + 8 * h + 16 * (r >> 2);
        f.u32[r] = *(const unsigned int*)(base + (row0 + n) * stride + k0 + kl);
    }
}

// B-fragment (32x16 bf16, K x N): lane n = lane&15, h = lane>>4.
// reg r holds K = 2*(r&3) + 8*h + 16*(r>>2) (+1). Source stored N-major [n][k] (i.e. transposed).
__device__ __forceinline__ void load_b_fragT(ABFrag& f, const bf16* baseT,
                                             int n0, int k0, int stride, int n, int h) {
#pragma unroll
    for (int r = 0; r < 8; ++r) {
        f.u32[r] = *(const unsigned int*)(baseT + (n0 + n) * stride + k0 + ((r & 3) * 2) + 8 * h + 16 * (r >> 2));
    }
}

__global__ __launch_bounds__(128)
void FlagOS_ChunkGatedDelta_kernel(const float* __restrict__ q,
                                   const float* __restrict__ k,
                                   const float* __restrict__ v,
                                   const float* __restrict__ beta,
                                   float* __restrict__ out) {
    __shared__ bf16 q_lds[BT * QS];      // Q chunk, bf16 row-major [i][d]
    __shared__ bf16 k_lds[BT * QS];      // K chunk, bf16 row-major [i][d]
    __shared__ bf16 kt_lds[DDIM * KTS];  // (K * dte)^T, bf16 [d][i]
    __shared__ bf16 vt_lds[EW * VTS];    // V slice transposed, bf16 [e][i]
    __shared__ bf16 a_lds[BT * AS];      // masked qk, bf16 [i][j]
    __shared__ bf16 stbf[EW * SBS];      // bf16 shadow of state, [e][d]
    __shared__ float St[EW * STS];       // state transposed f32 [e][d]
    __shared__ float cum[BT], dfs[BT], dte[BT], bsh[BT];
    __shared__ float exp_tot_sh;

    const int tid  = threadIdx.x;
    const int lane = tid & 31;
    const int wave = tid >> 5;
    const int n    = lane & 15;
    const int h    = lane >> 4;

    const int bh = blockIdx.x >> 2;   // which (b,h)
    const int es = blockIdx.x & 3;    // which e-slice
    const int e0 = es * EW;

    const float* qb = q    + (size_t)bh * NCH * BT * DDIM;
    const float* kb = k    + (size_t)bh * NCH * BT * DDIM;
    const float* vb = v    + (size_t)bh * NCH * BT * DDIM;
    const float* bb = beta + (size_t)bh * NCH * BT;
    float*       ob = out  + (size_t)bh * NCH * BT * DDIM;

    for (int idx = tid; idx < EW * STS; idx += 128) St[idx] = 0.0f;
    for (int idx = tid; idx < EW * SBS; idx += 128) stbf[idx] = (bf16)0.0f;
    __syncthreads();

    for (int c = 0; c < NCH; ++c) {
        const float* qc = qb + (size_t)c * BT * DDIM;
        const float* kc = kb + (size_t)c * BT * DDIM;
        const float* vc = vb + (size_t)c * BT * DDIM;
        float*       oc = ob + (size_t)c * BT * DDIM;

        // prefetch next chunk (global_prefetch_b8 path)
        if (c + 1 < NCH) {
            const float* qn = qb + (size_t)(c + 1) * BT * DDIM;
            const float* kn = kb + (size_t)(c + 1) * BT * DDIM;
            __builtin_prefetch(qn + tid * 64, 0, 0);
            __builtin_prefetch(kn + tid * 64, 0, 0);
        }

        // ---- gate math: cumsum(log(beta+eps)), decays ----
        if (tid < BT) bsh[tid] = bb[c * BT + tid];
        __syncthreads();
        if (tid == 0) {
            float s = 0.0f;
#pragma unroll 8
            for (int i = 0; i < BT; ++i) { s += __logf(bsh[i] + EPSF); cum[i] = s; }
        }
        __syncthreads();
        {
            float tot = cum[BT - 1];
            if (tid < BT) {
                dfs[tid] = __expf(cum[tid]);
                dte[tid] = __expf(tot - cum[tid]);
            }
            if (tid == 0) exp_tot_sh = __expf(tot);
        }

        // ---- stage Q, K (bf16, 4-wide: b128 global load -> b64 LDS store) ----
        for (int base = tid * 4; base < BT * DDIM; base += 128 * 4) {
            int i = base >> 7, d = base & 127;      // d is a multiple of 4
            float4 qv = *(const float4*)(qc + base);
            float4 kv = *(const float4*)(kc + base);
            Pack4 pq, pk;
            pq.h[0] = (bf16)qv.x; pq.h[1] = (bf16)qv.y;
            pq.h[2] = (bf16)qv.z; pq.h[3] = (bf16)qv.w;
            pk.h[0] = (bf16)kv.x; pk.h[1] = (bf16)kv.y;
            pk.h[2] = (bf16)kv.z; pk.h[3] = (bf16)kv.w;
            *(uint2*)(q_lds + i * QS + d) = pq.u2;
            *(uint2*)(k_lds + i * QS + d) = pk.u2;
        }
        // ---- stage transposed V slice (b128 global read, scattered b16 LDS writes) ----
        for (int base = tid * 4; base < BT * EW; base += 128 * 4) {
            int i = base >> 5, e = base & 31;       // e is a multiple of 4
            float4 vv = *(const float4*)(vc + i * DDIM + e0 + e);
            vt_lds[(e + 0) * VTS + i] = (bf16)vv.x;
            vt_lds[(e + 1) * VTS + i] = (bf16)vv.y;
            vt_lds[(e + 2) * VTS + i] = (bf16)vv.z;
            vt_lds[(e + 3) * VTS + i] = (bf16)vv.w;
        }
        __syncthreads();

        // ---- K^T scaled by dte (A-matrix for state update), b64 LDS stores ----
        for (int base = tid * 4; base < BT * DDIM; base += 128 * 4) {
            int d = base >> 6, i = base & 63;       // i is a multiple of 4
            Pack4 pk;
#pragma unroll
            for (int r = 0; r < 4; ++r)
                pk.h[r] = (bf16)((float)k_lds[(i + r) * QS + d] * dte[i + r]);
            *(uint2*)(kt_lds + d * KTS + i) = pk.u2;
        }
        __syncthreads();

        // ---- qk = Q K^T, causal decay mask, store bf16 A ----
#pragma unroll
        for (int tt = 0; tt < 4; ++tt) {
            int t  = wave * 4 + tt;
            int ti = t >> 2, tj = t & 3;
            f32x8 acc = {};
#pragma unroll
            for (int kk = 0; kk < 4; ++kk) {
                ABFrag a, b;
                load_a_frag(a, q_lds, ti * 16, kk * 32, QS, n, h);
                load_b_fragT(b, k_lds, tj * 16, kk * 32, QS, n, h);  // B[kd][j] = K[j][kd]
                acc = __builtin_amdgcn_wmma_f32_16x16x32_bf16(
                          false, a.bf, false, b.bf, (short)0, acc, false, false);
            }
            int j = tj * 16 + n;
#pragma unroll
            for (int r = 0; r < 8; ++r) {
                int i = ti * 16 + r + 8 * h;
                float val = (i >= j) ? acc[r] * __expf(cum[i] - cum[j]) : 0.0f;
                a_lds[i * AS + j] = (bf16)val;
            }
        }
        __syncthreads();

        // ---- out = dfs * (Q @ S) + (qk_masked @ V) ----
#pragma unroll
        for (int tt = 0; tt < 2; ++tt) {
            int t  = wave * 2 + tt;
            int ti = t >> 1, te = t & 1;
            f32x8 acc = {};
            // inter-chunk: Q @ S   (B from bf16 shadow of state)
#pragma unroll
            for (int kk = 0; kk < 4; ++kk) {
                ABFrag a, b;
                load_a_frag(a, q_lds, ti * 16, kk * 32, QS, n, h);
                load_b_fragT(b, stbf, te * 16, kk * 32, SBS, n, h);
                acc = __builtin_amdgcn_wmma_f32_16x16x32_bf16(
                          false, a.bf, false, b.bf, (short)0, acc, false, false);
            }
#pragma unroll
            for (int r = 0; r < 8; ++r) acc[r] *= dfs[ti * 16 + r + 8 * h];
            // intra-chunk: A @ V
#pragma unroll
            for (int kk = 0; kk < 2; ++kk) {
                ABFrag a, b;
                load_a_frag(a, a_lds, ti * 16, kk * 32, AS, n, h);
                load_b_fragT(b, vt_lds, te * 16, kk * 32, VTS, n, h);
                acc = __builtin_amdgcn_wmma_f32_16x16x32_bf16(
                          false, a.bf, false, b.bf, (short)0, acc, false, false);
            }
#pragma unroll
            for (int r = 0; r < 8; ++r) {
                oc[(size_t)(ti * 16 + r + 8 * h) * DDIM + e0 + te * 16 + n] = acc[r];
            }
        }
        __syncthreads();   // all inter reads of St/stbf done before update

        // ---- state update: S = S*exp(tot) + (dte*K)^T @ V ; refresh bf16 shadow ----
        {
            float etot = exp_tot_sh;
#pragma unroll
            for (int tt = 0; tt < 4; ++tt) {
                int t  = wave * 4 + tt;
                int td = t >> 1, te = t & 1;
                f32x8 acc = {};
#pragma unroll
                for (int kk = 0; kk < 2; ++kk) {
                    ABFrag a, b;
                    load_a_frag(a, kt_lds, td * 16, kk * 32, KTS, n, h);
                    load_b_fragT(b, vt_lds, te * 16, kk * 32, VTS, n, h);
                    acc = __builtin_amdgcn_wmma_f32_16x16x32_bf16(
                              false, a.bf, false, b.bf, (short)0, acc, false, false);
                }
#pragma unroll
                for (int r = 0; r < 8; ++r) {
                    int d = td * 16 + r + 8 * h;
                    int e = te * 16 + n;
                    float ns = St[e * STS + d] * etot + acc[r];
                    St[e * STS + d]  = ns;
                    stbf[e * SBS + d] = (bf16)ns;
                }
            }
        }
        __syncthreads();
    }
}

extern "C" void kernel_launch(void* const* d_in, const int* in_sizes, int n_in,
                              void* d_out, int out_size, void* d_ws, size_t ws_size,
                              hipStream_t stream) {
    const float* q    = (const float*)d_in[0];
    const float* k    = (const float*)d_in[1];
    const float* v    = (const float*)d_in[2];
    const float* beta = (const float*)d_in[3];
    float* out        = (float*)d_out;

    const int bh_count = in_sizes[3] / (NCH * BT);   // B*H = 32
    dim3 grid(bh_count * ESL);
    dim3 block(128);
    hipLaunchKernelGGL(FlagOS_ChunkGatedDelta_kernel, grid, block, 0, stream,
                       q, k, v, beta, out);
    (void)n_in; (void)out_size; (void)d_ws; (void)ws_size;
}
